// SAE_56281251447108
// MI455X (gfx1250) — matrix operand order repaired and checked by reading, not compiled, not run on hardware
//
#include <hip/hip_runtime.h>
#include <hip/hip_bf16.h>

// SAE forward: encoder GEMM (bf16 WMMA) -> ReLU -> exact top-64 threshold -> sparse decoder.
// Shapes: B=8192, D_EMB=1152, D_HID=32768, TOP_K=64.

#define SAE_B     8192
#define SAE_E     1152
#define SAE_H     32768
#define SAE_TOPK  64

typedef __attribute__((ext_vector_type(16))) __bf16        v16bf;
typedef __attribute__((ext_vector_type(8)))  float         v8f;
typedef __attribute__((ext_vector_type(4)))  unsigned int  u32x4;
typedef __attribute__((ext_vector_type(4)))  int           s32x4;
typedef __attribute__((ext_vector_type(4)))  float         f32x4;
typedef __attribute__((ext_vector_type(4)))  unsigned short u16x4;

// ---- CDNA5 async global->LDS path (ASYNCcnt), with compile-safe fallback ----
#if defined(__has_builtin)
#if __has_builtin(__builtin_amdgcn_global_load_async_to_lds_b128)
#define HAVE_ASYNC_LDS 1
#endif
#endif

#if defined(HAVE_ASYNC_LDS)
// Signature (from clang diagnostic): (v4i addrspace(1)*, v4i addrspace(3)*, imm offset, imm cpol)
#define GLD_ASYNC_B128(gp, lp)                                                 \
    __builtin_amdgcn_global_load_async_to_lds_b128(                            \
        (__attribute__((address_space(1))) s32x4*)(gp),                        \
        (__attribute__((address_space(3))) s32x4*)(lp), 0, 0)
#define WAIT_ASYNC0() asm volatile("s_wait_asynccnt 0x0" ::: "memory")
#endif

static __device__ __forceinline__ unsigned short f2bf(float f) {
    unsigned int u = __float_as_uint(f);
    unsigned int r = (u + 0x7FFFu + ((u >> 16) & 1u)) >> 16;   // RNE
    return (unsigned short)r;
}

// ---------------------------------------------------------------------------
// Kernel 1: f32 -> bf16 bulk convert (n multiple of 4)
// ---------------------------------------------------------------------------
__global__ __launch_bounds__(256)
void cvt_f32_bf16(const float* __restrict__ src, unsigned short* __restrict__ dst, int n) {
    int i = (blockIdx.x * blockDim.x + threadIdx.x) * 4;
    if (i < n) {
        f32x4 f = *(const f32x4*)(src + i);
        u16x4 h;
        h.x = f2bf(f.x); h.y = f2bf(f.y); h.z = f2bf(f.z); h.w = f2bf(f.w);
        *(u16x4*)(dst + i) = h;
    }
}

// ---------------------------------------------------------------------------
// Kernel 2: transpose W_down [1152, 32768] -> WdT [32768, 1152]
// block (32,8), grid (32768/32, 1152/32)
// ---------------------------------------------------------------------------
__global__ __launch_bounds__(256)
void transpose_wdown(const float* __restrict__ src, float* __restrict__ dst) {
    __shared__ float tile[32][33];
    int tx = threadIdx.x, ty = threadIdx.y;
    int bx = blockIdx.x * 32;   // column block in src (H dim)
    int by = blockIdx.y * 32;   // row block in src (E dim)
#pragma unroll
    for (int j = 0; j < 4; ++j) {
        int r = ty + j * 8;
        tile[r][tx] = src[(size_t)(by + r) * SAE_H + (bx + tx)];
    }
    __syncthreads();
#pragma unroll
    for (int j = 0; j < 4; ++j) {
        int r = ty + j * 8;   // dst row offset within block = src col offset
        dst[(size_t)(bx + r) * SAE_E + (by + tx)] = tile[tx][r];
    }
}

// ---------------------------------------------------------------------------
// Kernel 3: encoder GEMM  acts[B,H] = relu(embs[B,E] * W_up[H,E]^T + b_up)
// bf16 inputs, f32 accumulate via v_wmma_f32_16x16x32_bf16.
// Block tile 128x256, BK=32, 256 threads = 8 wave32 (2x4), wave tile 64x64
// (16 WMMAs per 16 ds_load_b128 fragment fetches per K-step).
// Tiles filled with GLOBAL_LOAD_ASYNC_TO_LDS_B128 (double-buffered, ASYNCcnt)
// when available; register-staged pipeline otherwise.
// ---------------------------------------------------------------------------
#define LDS_STRIDE 40   // ushort units; 80B row stride -> conflict-free b128 frag loads

__global__ __launch_bounds__(256)
void sae_encoder_gemm(const unsigned short* __restrict__ Abf,   // [B, E] bf16
                      const unsigned short* __restrict__ Wbf,   // [H, E] bf16
                      const float* __restrict__ b_up,           // [H]
                      float* __restrict__ acts)                 // [B, H]
{
    __shared__ __align__(16) unsigned short As[2][128 * LDS_STRIDE];
    __shared__ __align__(16) unsigned short Bs[2][256 * LDS_STRIDE];

    const int tid    = threadIdx.x;
    const int lane   = tid & 31;
    const int wave   = tid >> 5;
    const int wave_m = wave >> 2;   // 0..1 -> 64 rows each
    const int wave_n = wave & 3;    // 0..3 -> 64 cols each
    const int ml     = lane & 15;
    const int kh     = lane >> 4;   // k-half select per ISA A/B layouts

    const int m_base = blockIdx.y * 128;
    const int n_base = blockIdx.x * 256;

    // Tile fill decomposition: 16B segments; A = 512 segs (2/thread),
    // B = 1024 segs (4/thread). seg -> (row = trow + s*64, kseg = tks).
    const int trow = tid >> 2;
    const int tks  = (tid & 3) * 8;

    v8f c[4][4];
#pragma unroll
    for (int mi = 0; mi < 4; ++mi)
#pragma unroll
        for (int ni = 0; ni < 4; ++ni)
            c[mi][ni] = (v8f)(0.0f);

    const int KSTEPS = SAE_E / 32;   // 36

#if defined(HAVE_ASYNC_LDS)
    auto issue_tile = [&](int kk, int buf) {
        const int k0 = kk * 32 + tks;
#pragma unroll
        for (int s = 0; s < 2; ++s) {
            int r = trow + s * 64;
            GLD_ASYNC_B128(Abf + (size_t)(m_base + r) * SAE_E + k0,
                           &As[buf][r * LDS_STRIDE + tks]);
        }
#pragma unroll
        for (int s = 0; s < 4; ++s) {
            int r = trow + s * 64;
            GLD_ASYNC_B128(Wbf + (size_t)(n_base + r) * SAE_E + k0,
                           &Bs[buf][r * LDS_STRIDE + tks]);
        }
    };
    issue_tile(0, 0);
    WAIT_ASYNC0();
    __syncthreads();
#else
    u32x4 rg[6];
    auto preload = [&](int kk) {
        const int k0 = kk * 32 + tks;
        rg[0] = *(const u32x4*)(Abf + (size_t)(m_base + trow) * SAE_E + k0);
        rg[1] = *(const u32x4*)(Abf + (size_t)(m_base + trow + 64) * SAE_E + k0);
#pragma unroll
        for (int s = 0; s < 4; ++s)
            rg[2 + s] = *(const u32x4*)(Wbf + (size_t)(n_base + trow + s * 64) * SAE_E + k0);
    };
    preload(0);
#endif

    for (int kk = 0; kk < KSTEPS; ++kk) {
#if defined(HAVE_ASYNC_LDS)
        const int cur = kk & 1;
        if (kk + 1 < KSTEPS) issue_tile(kk + 1, cur ^ 1);
        const unsigned short* Asb = As[cur];
        const unsigned short* Bsb = Bs[cur];
#else
        if (kk > 0) __syncthreads();
        *(u32x4*)(&As[0][trow * LDS_STRIDE + tks])        = rg[0];
        *(u32x4*)(&As[0][(trow + 64) * LDS_STRIDE + tks]) = rg[1];
#pragma unroll
        for (int s = 0; s < 4; ++s)
            *(u32x4*)(&Bs[0][(trow + s * 64) * LDS_STRIDE + tks]) = rg[2 + s];
        __syncthreads();
        if (kk + 1 < KSTEPS) preload(kk + 1);
        const unsigned short* Asb = As[0];
        const unsigned short* Bsb = Bs[0];
#endif

        // Fragments per CDNA5 16-bit layouts:
        //  A 16x32: lane = M(0..15) + 16*khalf; VGPR0-3 k=kh*8+0..7, VGPR4-7 k=16+kh*8+0..7
        //  B 32x16: lane = N(0..15) + 16*khalf; VGPR0-3 k=kh*8+0..7, VGPR4-7 k=16+kh*8+0..7
        union Frag { u32x4 q[2]; v16bf v; };
        v16bf afrag[4];
#pragma unroll
        for (int mi = 0; mi < 4; ++mi) {
            int r = wave_m * 64 + mi * 16 + ml;
            Frag f;
            f.q[0] = *(const u32x4*)(Asb + r * LDS_STRIDE + kh * 8);
            f.q[1] = *(const u32x4*)(Asb + r * LDS_STRIDE + 16 + kh * 8);
            afrag[mi] = f.v;
        }
        v16bf bfrag[4];
#pragma unroll
        for (int ni = 0; ni < 4; ++ni) {
            int r = wave_n * 64 + ni * 16 + ml;
            Frag f;
            f.q[0] = *(const u32x4*)(Bsb + r * LDS_STRIDE + kh * 8);
            f.q[1] = *(const u32x4*)(Bsb + r * LDS_STRIDE + 16 + kh * 8);
            bfrag[ni] = f.v;
        }

#pragma unroll
        for (int mi = 0; mi < 4; ++mi)
#pragma unroll
            for (int ni = 0; ni < 4; ++ni)
                c[mi][ni] = __builtin_amdgcn_wmma_f32_16x16x32_bf16(
                    false, afrag[mi], false, bfrag[ni], (short)0, c[mi][ni], false, false);

#if defined(HAVE_ASYNC_LDS)
        WAIT_ASYNC0();       // next tile resident (this wave's share)
        __syncthreads();     // all waves done computing cur + all shares resident
#endif
    }

    // Epilogue: + b_up, ReLU, store.
    // C layout: lanes 0-15 -> M = r, lanes 16-31 -> M = r + 8; N = lane & 15.
#pragma unroll
    for (int ni = 0; ni < 4; ++ni) {
        int n = n_base + wave_n * 64 + ni * 16 + ml;
        float bu = b_up[n];
#pragma unroll
        for (int mi = 0; mi < 4; ++mi) {
            int mrow0 = m_base + wave_m * 64 + mi * 16 + kh * 8;
#pragma unroll
            for (int r = 0; r < 8; ++r) {
                float v = c[mi][ni][r] + bu;
                acts[(size_t)(mrow0 + r) * SAE_H + n] = v > 0.0f ? v : 0.0f;
            }
        }
    }
}

// ---------------------------------------------------------------------------
// Kernel 4: per-row exact 65th-largest via 4-pass radix select in LDS,
// compact actives, sparse decoder  out[b,:] = b_down + sum val_j * WdT[h_j,:]
// One block (256 threads) per row; dynamic LDS = 32768 floats (128 KB).
// ---------------------------------------------------------------------------
__global__ __launch_bounds__(256)
void sae_topk_decode(const float* __restrict__ acts,     // [B, H]
                     const float* __restrict__ WdT,      // [H, E]
                     const float* __restrict__ b_down,   // [E]
                     float* __restrict__ out)            // [B, E]
{
    extern __shared__ float sx[];                        // SAE_H floats
    __shared__ unsigned int hist[256];
    __shared__ float svals[128];
    __shared__ int   sidx[128];
    __shared__ unsigned int s_prefix;
    __shared__ int s_k;
    __shared__ int s_nact;

    const int row = blockIdx.x;
    const int tid = threadIdx.x;
    const float* arow = acts + (size_t)row * SAE_H;

#if defined(HAVE_ASYNC_LDS)
    for (int i = tid * 4; i < SAE_H; i += 256 * 4)
        GLD_ASYNC_B128(arow + i, sx + i);
#else
    for (int i = tid; i < SAE_H; i += 256) sx[i] = arow[i];
#endif
    if (tid == 0) { s_prefix = 0u; s_k = SAE_TOPK + 1; s_nact = 0; }
#if defined(HAVE_ASYNC_LDS)
    WAIT_ASYNC0();
#endif
    __syncthreads();

    // Radix select: non-negative (post-ReLU) floats are monotonic as uint32.
    for (int pass = 3; pass >= 0; --pass) {
        const int shift = pass * 8;
        hist[tid] = 0u;
        __syncthreads();
        const unsigned int pref = s_prefix;
        for (int i = tid; i < SAE_H; i += 256) {
            unsigned int u = __float_as_uint(sx[i]);
            bool match = (pass == 3) || ((u >> (shift + 8)) == (pref >> (shift + 8)));
            if (match) atomicAdd(&hist[(u >> shift) & 255u], 1u);
        }
        __syncthreads();
        if (tid == 0) {
            int kk = s_k;
            unsigned int cum = 0;
            int b = 255;
            for (; b >= 0; --b) { cum += hist[b]; if ((int)cum >= kk) break; }
            if (b < 0) b = 0;                       // defensive; invariant keeps cum>=kk
            s_k = kk - (int)(cum - hist[b]);        // rank within selected bin
            s_prefix = pref | ((unsigned int)b << shift);
        }
        __syncthreads();
    }
    const float thr = __uint_as_float(s_prefix);    // exact 65th-largest value

    // Compact strictly-greater entries (<= 64 by construction).
    for (int i = tid; i < SAE_H; i += 256) {
        float v = sx[i];
        if (v > thr) {
            int p = atomicAdd(&s_nact, 1);
            if (p < 128) { svals[p] = v; sidx[p] = i; }
        }
    }
    __syncthreads();
    int m = s_nact; if (m > 128) m = 128;

    // Sparse decoder; WdT rows are contiguous -> coalesced, L2-resident (151 MB < 192 MB L2).
    for (int e = tid; e < SAE_E; e += 256) {
        float acc = b_down[e];
        for (int j = 0; j < m; ++j)
            acc += svals[j] * WdT[(size_t)sidx[j] * SAE_E + e];
        out[(size_t)row * SAE_E + e] = acc;
    }
}

// ---------------------------------------------------------------------------
// Host entry
// ---------------------------------------------------------------------------
extern "C" void kernel_launch(void* const* d_in, const int* in_sizes, int n_in,
                              void* d_out, int out_size, void* d_ws, size_t ws_size,
                              hipStream_t stream) {
    (void)in_sizes; (void)n_in; (void)out_size; (void)ws_size;

    const float* embs   = (const float*)d_in[0];   // [B, E]
    const float* W_up   = (const float*)d_in[1];   // [H, E]
    const float* b_up   = (const float*)d_in[2];   // [H]
    const float* W_down = (const float*)d_in[3];   // [E, H]
    const float* b_down = (const float*)d_in[4];   // [E]
    float* out = (float*)d_out;                    // [B, E]

    char* ws = (char*)d_ws;
    size_t off = 0;
    float* acts = (float*)(ws + off);          off += (size_t)SAE_B * SAE_H * sizeof(float);
    unsigned short* Abf = (unsigned short*)(ws + off); off += (size_t)SAE_B * SAE_E * sizeof(unsigned short);
    unsigned short* Wbf = (unsigned short*)(ws + off); off += (size_t)SAE_H * SAE_E * sizeof(unsigned short);
    float* WdT = (float*)(ws + off);           off += (size_t)SAE_H * SAE_E * sizeof(float);

    // 1) convert embs and W_up to bf16
    {
        int nA = SAE_B * SAE_E;
        cvt_f32_bf16<<<(nA / 4 + 255) / 256, 256, 0, stream>>>(embs, Abf, nA);
        int nW = SAE_H * SAE_E;
        cvt_f32_bf16<<<(nW / 4 + 255) / 256, 256, 0, stream>>>(W_up, Wbf, nW);
    }

    // 2) transpose W_down -> WdT [H, E]
    {
        dim3 blk(32, 8), grd(SAE_H / 32, SAE_E / 32);
        transpose_wdown<<<grd, blk, 0, stream>>>(W_down, WdT);
    }

    // 3) encoder GEMM with bf16 WMMA (block tile 128x256)
    {
        dim3 grd(SAE_H / 256, SAE_B / 128);   // (128, 64)
        sae_encoder_gemm<<<grd, 256, 0, stream>>>(Abf, Wbf, b_up, acts);
    }

    // 4) top-k threshold + sparse decoder (128 KB dynamic LDS per block)
    {
        size_t smem = (size_t)SAE_H * sizeof(float);
        sae_topk_decode<<<SAE_B, 256, smem, stream>>>(acts, WdT, b_down, out);
    }
}
